// BasicLaplacePINN_3367254360719
// MI455X (gfx1250) — matrix operand order repaired and checked by reading, not compiled, not run on hardware
//
#include <hip/hip_runtime.h>
#include <hip/hip_bf16.h>
#include <math.h>

// Divergence of sigma * grad(MLP) == sigma * Laplacian(MLP), via forward
// 2nd-order propagation of (h, dh_i, d2h_i) i=0..2 through the net.
// One wave32 = 16 samples. 64x64 layers done with v_wmma_f32_16x16x32_f16,
// computing Z^T = W^T @ H^T so the sample index stays in lane&15 across
// B-operand and C/D layouts (only a lane<->lane+16 half swap is needed).

typedef __attribute__((ext_vector_type(16))) _Float16 v16h;
typedef __attribute__((ext_vector_type(8)))  _Float16 v8h;
typedef __attribute__((ext_vector_type(8)))  float    v8f;

#define HID 64
#define PAD 72   // padded LDS row length in halves (bank-conflict-free b128 loads)

union HF16 { v16h v; v8h h[2]; _Float16 e[16]; };
union HF8  { v8h v;  _Float16 e[8]; };
union FF8  { v8f v;  float e[8]; };

__device__ __forceinline__ v8f vzero8() {
  v8f z = {0.f,0.f,0.f,0.f,0.f,0.f,0.f,0.f};
  return z;
}

// Branch-free tanh on the hardware transcendentals (v_exp_f32 / v_rcp_f32).
// tanh(z) = 1 - 2/(exp2(2*log2e*z) + 1); saturates correctly at +-1.
__device__ __forceinline__ float fast_tanh(float z) {
  float e = __builtin_amdgcn_exp2f(z * 2.8853900817779268f);
  return 1.0f - 2.0f * __builtin_amdgcn_rcpf(e + 1.0f);
}

__device__ __forceinline__ v8f wmma16(v16h a, v16h b, v8f c) {
  // D = A(16x32 f16) * B(32x16 f16) + C(16x16 f32)
  return __builtin_amdgcn_wmma_f32_16x16x32_f16(false, a, false, b, (short)0, c,
                                                false, false);
}

// A-operand fragment of W^T (row-major in LDS, stride PAD halves):
// rows m = 16*mt + (lane&15); k-slice = 32*kt; per-lane halves:
//   elems 0..7  -> k = 32*kt + 16*hi + 0..7
//   elems 8..15 -> k = 32*kt + 16*hi + 8 + 0..7
__device__ __forceinline__ v16h load_a(const _Float16* wt, int mt, int kt,
                                       int lane15, int hi) {
  const _Float16* p = wt + (mt * 16 + lane15) * PAD + kt * 32 + hi * 16;
  HF16 a;
  a.h[0] = *(const v8h*)(p);
  a.h[1] = *(const v8h*)(p + 8);
  return a.v;
}

// acc[mt] += W^T @ G  (W^T 64x64 from LDS, G = 2 B-fragments of 64x16).
// Back-to-back calls with the same wt reuse the LDS loads via compiler CSE.
__device__ __forceinline__ void mm64(v8f acc[4], const HF16 g[2],
                                     const _Float16* wt, int lane15, int hi) {
#pragma unroll
  for (int mt = 0; mt < 4; ++mt) {
    acc[mt] = wmma16(load_a(wt, mt, 0, lane15, hi), g[0].v, acc[mt]);
    acc[mt] = wmma16(load_a(wt, mt, 1, lane15, hi), g[1].v, acc[mt]);
  }
}

// D-layout f32 tiles (4 x v8f, hidden = 16*mt + 8*hi + r, sample = lane&15)
//   -> B-layout f16 fragments (k = 32*kt + 16*hi + j, sample = lane&15).
// Needs only a lane <-> lane^16 exchange for half of the values.
__device__ __forceinline__ void d2b(const v8f t[4], HF16 frag[2], int hi) {
#pragma unroll
  for (int kt = 0; kt < 2; ++kt) {
    FF8 a0, a1, s0, s1;
    a0.v = t[2 * kt];
    a1.v = t[2 * kt + 1];
#pragma unroll
    for (int r = 0; r < 8; ++r) {
      s0.e[r] = __shfl_xor(a0.e[r], 16, 32);
      s1.e[r] = __shfl_xor(a1.e[r], 16, 32);
    }
#pragma unroll
    for (int r = 0; r < 8; ++r) {
      float lo = hi ? s1.e[r] : a0.e[r];   // elems j=0..7
      float hh = hi ? a1.e[r] : s0.e[r];   // elems j=8..15
      frag[kt].e[r]     = (_Float16)lo;
      frag[kt].e[r + 8] = (_Float16)hh;
    }
  }
}

__global__ __launch_bounds__(256, 1) void BasicLaplacePINN_kernel(
    const float* __restrict__ x, const float* __restrict__ sigma,
    const float* __restrict__ W1, const float* __restrict__ b1,
    const float* __restrict__ W2, const float* __restrict__ b2,
    const float* __restrict__ W3, const float* __restrict__ b3,
    const float* __restrict__ W4, float* __restrict__ out, int nB) {
  __shared__ _Float16 sWT2[HID * PAD];
  __shared__ _Float16 sWT3[HID * PAD];
  __shared__ float sW1[3 * HID];
  __shared__ float sb1[HID], sb2[HID], sb3[HID], sW4[HID];

  // Stage weights: W2/W3 transposed to f16 (WT[n][k] = W[k][n]), rest f32.
  for (int t = threadIdx.x; t < HID * HID; t += blockDim.x) {
    int k = t >> 6, n = t & 63;
    sWT2[n * PAD + k] = (_Float16)W2[t];
    sWT3[n * PAD + k] = (_Float16)W3[t];
  }
  for (int t = threadIdx.x; t < 3 * HID; t += blockDim.x) sW1[t] = W1[t];
  for (int t = threadIdx.x; t < HID; t += blockDim.x) {
    sb1[t] = b1[t]; sb2[t] = b2[t]; sb3[t] = b3[t]; sW4[t] = W4[t];
  }
  __syncthreads();

  const int lane   = threadIdx.x & 31;
  const int lane15 = lane & 15;
  const int hi     = lane >> 4;
  const int tile   = blockIdx.x * (blockDim.x >> 5) + (threadIdx.x >> 5);
  const int base   = tile * 16;
  if (base >= nB) return;               // wave-uniform: EXEC stays all-ones

  const int   sidx = base + lane15;
  const float x0 = x[sidx * 3 + 0];
  const float x1 = x[sidx * 3 + 1];
  const float x2 = x[sidx * 3 + 2];
  const float sg = sigma[sidx];

  // State: B-layout f16 fragments (sample = lane&15).
  HF16 fh[2], fdh[3][2], f2h[3][2];

  // ---------- layer 1: 3 -> 64, elementwise directly into B layout ----------
#pragma unroll
  for (int kt = 0; kt < 2; ++kt) {
#pragma unroll
    for (int j = 0; j < 16; ++j) {
      int k = kt * 32 + hi * 16 + j;
      float w0 = sW1[k], w1 = sW1[64 + k], w2 = sW1[128 + k];
      float z = x0 * w0 + x1 * w1 + x2 * w2 + sb1[k];
      float t  = fast_tanh(z);
      float sc = 1.0f - t * t;
      float uc = -2.0f * t * sc;
      fh[kt].e[j]     = (_Float16)t;
      fdh[0][kt].e[j] = (_Float16)(sc * w0);
      fdh[1][kt].e[j] = (_Float16)(sc * w1);
      fdh[2][kt].e[j] = (_Float16)(sc * w2);
      f2h[0][kt].e[j] = (_Float16)(uc * w0 * w0);
      f2h[1][kt].e[j] = (_Float16)(uc * w1 * w1);
      f2h[2][kt].e[j] = (_Float16)(uc * w2 * w2);
    }
  }

  // ---------- layer 2: 64 -> 64 (WMMA) ----------
  {
    v8f zt[4];
#pragma unroll
    for (int mt = 0; mt < 4; ++mt) {
      FF8 z;
#pragma unroll
      for (int r = 0; r < 8; ++r) z.e[r] = sb2[mt * 16 + hi * 8 + r];
      zt[mt] = z.v;
    }
    mm64(zt, fh, sWT2, lane15, hi);

    HF8 sh[4], uh[4];                   // s = 1-t^2, u = -2 t s (packed f16)
#pragma unroll
    for (int mt = 0; mt < 4; ++mt) {
      FF8 z; z.v = zt[mt];
#pragma unroll
      for (int r = 0; r < 8; ++r) {
        float t  = fast_tanh(z.e[r]);
        float sc = 1.0f - t * t;
        z.e[r] = t;
        sh[mt].e[r] = (_Float16)sc;
        uh[mt].e[r] = (_Float16)(-2.0f * t * sc);
      }
      zt[mt] = z.v;
    }
    d2b(zt, fh, hi);                    // new h fragments

#pragma unroll
    for (int i = 0; i < 3; ++i) {
      v8f dz[4], d2[4];
#pragma unroll
      for (int mt = 0; mt < 4; ++mt) { dz[mt] = vzero8(); d2[mt] = vzero8(); }
      mm64(dz, fdh[i], sWT2, lane15, hi);
      mm64(d2, f2h[i], sWT2, lane15, hi);
#pragma unroll
      for (int mt = 0; mt < 4; ++mt) {
        FF8 a, b; a.v = dz[mt]; b.v = d2[mt];
#pragma unroll
        for (int r = 0; r < 8; ++r) {
          float s  = (float)sh[mt].e[r];
          float u  = (float)uh[mt].e[r];
          float av = a.e[r];
          b.e[r] = s * b.e[r] + u * av * av;   // d2a = s*d2z + u*dz^2
          a.e[r] = s * av;                     // da  = s*dz
        }
        dz[mt] = a.v; d2[mt] = b.v;
      }
      d2b(dz, fdh[i], hi);
      d2b(d2, f2h[i], hi);
    }
  }

  // ---------- layer 3 (WMMA) + final linear 64 -> 1 ----------
  float lapsum = 0.f;
  {
    v8f zt[4];
#pragma unroll
    for (int mt = 0; mt < 4; ++mt) {
      FF8 z;
#pragma unroll
      for (int r = 0; r < 8; ++r) z.e[r] = sb3[mt * 16 + hi * 8 + r];
      zt[mt] = z.v;
    }
    mm64(zt, fh, sWT3, lane15, hi);

    HF8 sh[4], uh[4];
#pragma unroll
    for (int mt = 0; mt < 4; ++mt) {
      FF8 z; z.v = zt[mt];
#pragma unroll
      for (int r = 0; r < 8; ++r) {
        float t  = fast_tanh(z.e[r]);
        float sc = 1.0f - t * t;
        sh[mt].e[r] = (_Float16)sc;
        uh[mt].e[r] = (_Float16)(-2.0f * t * sc);
      }
    }

#pragma unroll
    for (int i = 0; i < 3; ++i) {
      v8f dz[4], d2[4];
#pragma unroll
      for (int mt = 0; mt < 4; ++mt) { dz[mt] = vzero8(); d2[mt] = vzero8(); }
      mm64(dz, fdh[i], sWT3, lane15, hi);
      mm64(d2, f2h[i], sWT3, lane15, hi);
      // lapsum += sum_m W4[m] * (s*d2z + u*dz^2)  -- reduced in D layout
#pragma unroll
      for (int mt = 0; mt < 4; ++mt) {
        FF8 a, b; a.v = dz[mt]; b.v = d2[mt];
#pragma unroll
        for (int r = 0; r < 8; ++r) {
          float s  = (float)sh[mt].e[r];
          float u  = (float)uh[mt].e[r];
          float av = a.e[r];
          float d2a = s * b.e[r] + u * av * av;
          lapsum += sW4[mt * 16 + hi * 8 + r] * d2a;
        }
      }
    }
  }

  // Lanes L and L+16 hold complementary hidden halves of the same sample.
  float tot = lapsum + __shfl_xor(lapsum, 16, 32);
  if (hi == 0) out[sidx] = sg * tot;
}

extern "C" void kernel_launch(void* const* d_in, const int* in_sizes, int n_in,
                              void* d_out, int out_size, void* d_ws,
                              size_t ws_size, hipStream_t stream) {
  const float* x  = (const float*)d_in[0];
  const float* sg = (const float*)d_in[1];
  const float* W1 = (const float*)d_in[2];
  const float* b1 = (const float*)d_in[3];
  const float* W2 = (const float*)d_in[4];
  const float* b2 = (const float*)d_in[5];
  const float* W3 = (const float*)d_in[6];
  const float* b3 = (const float*)d_in[7];
  const float* W4 = (const float*)d_in[8];
  // d_in[9] = b4: constant offset, vanishes under differentiation.

  const int nB    = in_sizes[1];          // sigma has B elements
  const int tiles = (nB + 15) / 16;       // 16 samples per wave
  const int waves = 8;                    // 256 threads = 8 wave32
  const int blocks = (tiles + waves - 1) / waves;

  BasicLaplacePINN_kernel<<<blocks, 256, 0, stream>>>(
      x, sg, W1, b1, W2, b2, W3, b3, W4, (float*)d_out, nB);
}
